// ACTSetTransformer_26963804684805
// MI455X (gfx1250) — matrix operand order repaired and checked by reading, not compiled, not run on hardware
//
#include <hip/hip_runtime.h>
#include <math.h>

// ---------------- model constants (from reference) ----------------
static constexpr int D     = 128;
static constexpr int HEADS = 4;
static constexpr int DH    = D / HEADS;   // 32
static constexpr int BATCH = 64;
static constexpr int NTOK  = 2048;
static constexpr int NI    = 32;          // NUM_INDS
static constexpr int KMAX  = 6;
static constexpr int DX    = 2;
static constexpr int OUTC  = 6;           // 2 + DIM_PARAMS

typedef __attribute__((ext_vector_type(16))) _Float16 v16h;
typedef __attribute__((ext_vector_type(8)))  _Float16 v8h;
typedef __attribute__((ext_vector_type(8)))  float    v8f;

__device__ __forceinline__ v8f wmma_f16(v16h a, v16h b, v8f c) {
  return __builtin_amdgcn_wmma_f32_16x16x32_f16(
      false, a, false, b, (short)0, c, false, false);
}

__device__ __forceinline__ float softplusf(float x) {
  return fmaxf(x, 0.f) + log1pf(__expf(-fabsf(x)));
}

__device__ __forceinline__ void pack4(v16h& a, int base, float4 v) {
  a[base + 0] = (_Float16)v.x;
  a[base + 1] = (_Float16)v.y;
  a[base + 2] = (_Float16)v.z;
  a[base + 3] = (_Float16)v.w;
}

// B-fragment from fragment-native f16 weights: 32 contiguous bytes.
__device__ __forceinline__ v16h load_bfrag(const _Float16* __restrict__ base) {
  const v8h b0 = *(const v8h*)(base);
  const v8h b1 = *(const v8h*)(base + 8);
  v16h b;
#pragma unroll
  for (int i = 0; i < 8; ++i) { b[i] = b0[i]; b[8 + i] = b1[i]; }
  return b;
}

// A-fragment (16x32 f16) from a row-major f32 row, 16 contiguous k from kb (16-aligned).
__device__ __forceinline__ v16h load_afrag(const float* __restrict__ row, int kb) {
  v16h a;
  pack4(a, 0, *(const float4*)(row + kb));
  pack4(a, 4, *(const float4*)(row + kb + 4));
  pack4(a, 8, *(const float4*)(row + kb + 8));
  pack4(a, 12, *(const float4*)(row + kb + 12));
  return a;
}

// =====================================================================
// Weight prep: W (KxN f32) -> WF fragment-native f16, K zero-padded to 32
// WF[((k>>5)*N + c)*32 + (k&31)]
// =====================================================================
__global__ void k_wprep(const float* __restrict__ W, _Float16* __restrict__ WF,
                        int K, int N, int total /* Kpad*N */) {
  const int i = blockIdx.x * blockDim.x + threadIdx.x;
  if (i >= total) return;
  const int k = i / N, c = i % N;
  const float v = (k < K) ? W[(size_t)k * N + c] : 0.f;
  WF[(((size_t)(k >> 5) * N + c) << 5) + (k & 31)] = (_Float16)v;
}

// =====================================================================
// K=128 specialized GEMM: M%16==0, N%64==0. 16x64 strip per wave,
// fully unrolled 4 k-steps; fragments loaded ahead of the WMMA bursts.
// =====================================================================
__global__ void __launch_bounds__(32)
k_gemm_f128(const float* __restrict__ X, const _Float16* __restrict__ WF,
            const float* __restrict__ bias, const float* __restrict__ resid,
            float* __restrict__ Y, int M, int N, int op) {
  const int lane = threadIdx.x & 31;
  const int half = lane >> 4;
  const int row0 = blockIdx.x << 4;
  const int col0 = blockIdx.y << 6;
  const float* Arow = X + (size_t)(row0 + (lane & 15)) * D;
  const int c0 = col0 + (lane & 15);

  // All four A-fragments up front (16 x b128 loads).
  v16h afr[4];
#pragma unroll
  for (int s = 0; s < 4; ++s) afr[s] = load_afrag(Arow, s * 32 + (half << 4));

  v8f acc[4];
  const v8f zero = {0.f, 0.f, 0.f, 0.f, 0.f, 0.f, 0.f, 0.f};
#pragma unroll
  for (int t = 0; t < 4; ++t) acc[t] = zero;

#pragma unroll
  for (int s = 0; s < 4; ++s) {
    const _Float16* wb = WF + (((size_t)s * N + c0) << 5) + (half << 4);
    v16h bfr[4];
#pragma unroll
    for (int t = 0; t < 4; ++t) bfr[t] = load_bfrag(wb + t * 512);
#pragma unroll
    for (int t = 0; t < 4; ++t) acc[t] = wmma_f16(afr[s], bfr[t], acc[t]);
  }

#pragma unroll
  for (int t = 0; t < 4; ++t) {
    const int c = col0 + t * 16 + (lane & 15);
    const float bv = bias[c];
#pragma unroll
    for (int v = 0; v < 8; ++v) {
      const int r = row0 + (half << 3) + v;
      float tv = acc[t][v] + bv;
      if (op == 1) tv = resid[(size_t)r * N + c] + fmaxf(tv, 0.f);
      Y[(size_t)r * N + c] = tv;
    }
  }
}

// Runtime-K fast GEMM: M%16==0, K%32==0, N%64==0.
__global__ void __launch_bounds__(32)
k_gemm_f(const float* __restrict__ X, const _Float16* __restrict__ WF,
         const float* __restrict__ bias, const float* __restrict__ resid,
         float* __restrict__ Y, int M, int N, int K, int op) {
  const int lane = threadIdx.x & 31;
  const int half = lane >> 4;
  const int row0 = blockIdx.x << 4;
  const int col0 = blockIdx.y << 6;
  const float* Arow = X + (size_t)(row0 + (lane & 15)) * K;
  const int c0 = col0 + (lane & 15);

  v8f acc[4];
  const v8f zero = {0.f, 0.f, 0.f, 0.f, 0.f, 0.f, 0.f, 0.f};
#pragma unroll
  for (int t = 0; t < 4; ++t) acc[t] = zero;

  for (int k0 = 0; k0 < K; k0 += 32) {
    const v16h a = load_afrag(Arow, k0 + (half << 4));
    const _Float16* wb = WF + (((size_t)(k0 >> 5) * N + c0) << 5) + (half << 4);
    v16h bfr[4];
#pragma unroll
    for (int t = 0; t < 4; ++t) bfr[t] = load_bfrag(wb + t * 512);
#pragma unroll
    for (int t = 0; t < 4; ++t) acc[t] = wmma_f16(a, bfr[t], acc[t]);
  }

#pragma unroll
  for (int t = 0; t < 4; ++t) {
    const int c = col0 + t * 16 + (lane & 15);
    const float bv = bias[c];
#pragma unroll
    for (int v = 0; v < 8; ++v) {
      const int r = row0 + (half << 3) + v;
      float tv = acc[t][v] + bv;
      if (op == 1) tv = resid[(size_t)r * N + c] + fmaxf(tv, 0.f);
      Y[(size_t)r * N + c] = tv;
    }
  }
}

// EDGE GEMM: any M,N,K — branchless clamp+select, guarded stores.
__global__ void __launch_bounds__(32)
k_gemm_e(const float* __restrict__ X, const _Float16* __restrict__ WF,
         const float* __restrict__ bias, const float* __restrict__ resid,
         float* __restrict__ Y, int M, int N, int K, int op) {
  const int lane = threadIdx.x & 31;
  const int half = lane >> 4;
  const int row0 = blockIdx.x << 4;
  const int col0 = blockIdx.y << 4;
  const int ar = row0 + (lane & 15);
  const bool arok = ar < M;
  const float* Arow = X + (size_t)(arok ? ar : (M - 1)) * K;
  const int c = col0 + (lane & 15);
  const int cc = (c < N) ? c : (N - 1);

  v8f acc = {0.f, 0.f, 0.f, 0.f, 0.f, 0.f, 0.f, 0.f};
  for (int k0 = 0; k0 < K; k0 += 32) {
    v16h a;
    const int kbA = k0 + (half << 4);
#pragma unroll
    for (int i = 0; i < 8; ++i) {
      const int k1 = kbA + i, k2 = kbA + 8 + i;
      const float v1 = Arow[(k1 < K) ? k1 : (K - 1)];   // unconditional load
      const float v2 = Arow[(k2 < K) ? k2 : (K - 1)];
      a[i]     = (_Float16)((arok && k1 < K) ? v1 : 0.f);  // value select only
      a[8 + i] = (_Float16)((arok && k2 < K) ? v2 : 0.f);
    }
    const _Float16* wb = WF + (((size_t)(k0 >> 5) * N + cc) << 5) + (half << 4);
    acc = wmma_f16(a, load_bfrag(wb), acc);   // WF K-padded with zeros
  }
  if (c >= N) return;
  const float bv = bias[c];
#pragma unroll
  for (int v = 0; v < 8; ++v) {
    const int r = row0 + (half << 3) + v;
    if (r < M) {
      float t = acc[v] + bv;
      if (op == 1) t = resid[(size_t)r * N + c] + fmaxf(t, 0.f);
      Y[(size_t)r * N + c] = t;
    }
  }
}

// =====================================================================
// Scores: S = Qh @ Kh^T * scale (dh=32 -> single WMMA)
// =====================================================================
__global__ void __launch_bounds__(32)
k_scores_f(const float* __restrict__ Q, const float* __restrict__ Km,
           float* __restrict__ S, int nq, int nk, float scale) {
  const int lane = threadIdx.x & 31;
  const int half = lane >> 4;
  const int bh = blockIdx.z;
  const int b = bh / HEADS, h = bh % HEADS;
  const int row0 = blockIdx.x << 4;
  const int col0 = blockIdx.y << 4;
  const float* Qb = Q + (size_t)b * nq * D + h * DH;
  const float* Kb = Km + (size_t)b * nk * D + h * DH;

  const v16h a = load_afrag(Qb + (size_t)(row0 + (lane & 15)) * D, half << 4);
  v16h bf;
  {
    const float* src = Kb + (size_t)(col0 + (lane & 15)) * D + (half << 4);
    pack4(bf, 0, *(const float4*)(src));
    pack4(bf, 4, *(const float4*)(src + 4));
    pack4(bf, 8, *(const float4*)(src + 8));
    pack4(bf, 12, *(const float4*)(src + 12));
  }
  v8f acc = {0.f, 0.f, 0.f, 0.f, 0.f, 0.f, 0.f, 0.f};
  acc = wmma_f16(a, bf, acc);

  float* Sb = S + (size_t)bh * nq * nk;
  const int c = col0 + (lane & 15);
#pragma unroll
  for (int v = 0; v < 8; ++v) {
    const int r = row0 + (half << 3) + v;
    Sb[(size_t)r * nk + c] = acc[v] * scale;
  }
}

// Clamped scores: any nq/nk (loads clamped, stores guarded)
__global__ void __launch_bounds__(32)
k_scores_c(const float* __restrict__ Q, const float* __restrict__ Km,
           float* __restrict__ S, int nq, int nk, float scale) {
  const int lane = threadIdx.x & 31;
  const int half = lane >> 4;
  const int bh = blockIdx.z;
  const int b = bh / HEADS, h = bh % HEADS;
  const int row0 = blockIdx.x << 4;
  const int col0 = blockIdx.y << 4;
  const float* Qb = Q + (size_t)b * nq * D + h * DH;
  const float* Kb = Km + (size_t)b * nk * D + h * DH;
  const int rA = row0 + (lane & 15);
  const int nB = col0 + (lane & 15);

  const v16h a = load_afrag(Qb + (size_t)((rA < nq) ? rA : (nq - 1)) * D, half << 4);
  v16h bf;
  {
    const float* src = Kb + (size_t)((nB < nk) ? nB : (nk - 1)) * D + (half << 4);
    pack4(bf, 0, *(const float4*)(src));
    pack4(bf, 4, *(const float4*)(src + 4));
    pack4(bf, 8, *(const float4*)(src + 8));
    pack4(bf, 12, *(const float4*)(src + 12));
  }
  v8f acc = {0.f, 0.f, 0.f, 0.f, 0.f, 0.f, 0.f, 0.f};
  acc = wmma_f16(a, bf, acc);

  float* Sb = S + (size_t)bh * nq * nk;
  const int c = col0 + (lane & 15);
  if (c >= nk) return;
#pragma unroll
  for (int v = 0; v < 8; ++v) {
    const int r = row0 + (half << 3) + v;
    if (r < nq) Sb[(size_t)r * nk + c] = acc[v] * scale;
  }
}

// =====================================================================
// O = Qh + P @ Vh
// =====================================================================
__global__ void __launch_bounds__(32)
k_av_f(const float* __restrict__ P, const float* __restrict__ V,
       const float* __restrict__ Q, float* __restrict__ O, int nq, int nk) {
  const int lane = threadIdx.x & 31;
  const int half = lane >> 4;
  const int bh = blockIdx.z;
  const int b = bh / HEADS, h = bh % HEADS;
  const int row0 = blockIdx.x << 4;
  const int col0 = blockIdx.y << 4;  // within DH
  const float* Prow = P + (size_t)bh * nq * nk + (size_t)(row0 + (lane & 15)) * nk;
  const float* Vcol = V + (size_t)b * nk * D + h * DH + col0 + (lane & 15);

  v8f acc = {0.f, 0.f, 0.f, 0.f, 0.f, 0.f, 0.f, 0.f};
#pragma unroll 2
  for (int k0 = 0; k0 < nk; k0 += 32) {
    const v16h a = load_afrag(Prow, k0 + (half << 4));
    const int kbB = k0 + (half << 4);
    v16h bf;
#pragma unroll
    for (int i = 0; i < 16; ++i) bf[i] = (_Float16)Vcol[(size_t)(kbB + i) * D];
    acc = wmma_f16(a, bf, acc);
  }
  const float* Qb = Q + (size_t)b * nq * D + h * DH + col0 + (lane & 15);
  float* Ob = O + (size_t)b * nq * D + h * DH + col0 + (lane & 15);
#pragma unroll
  for (int v = 0; v < 8; ++v) {
    const size_t r = (size_t)(row0 + (half << 3) + v) * D;
    Ob[r] = Qb[r] + acc[v];
  }
}

// Clamped AV: any nq/nk — branchless clamp+select, guarded stores.
__global__ void __launch_bounds__(32)
k_av_c(const float* __restrict__ P, const float* __restrict__ V,
       const float* __restrict__ Q, float* __restrict__ O, int nq, int nk) {
  const int lane = threadIdx.x & 31;
  const int half = lane >> 4;
  const int bh = blockIdx.z;
  const int b = bh / HEADS, h = bh % HEADS;
  const int row0 = blockIdx.x << 4;
  const int col0 = blockIdx.y << 4;
  const int ar = row0 + (lane & 15);
  const float* Prow = P + (size_t)bh * nq * nk + (size_t)((ar < nq) ? ar : (nq - 1)) * nk;
  const float* Vcol = V + (size_t)b * nk * D + h * DH + col0 + (lane & 15);

  v8f acc = {0.f, 0.f, 0.f, 0.f, 0.f, 0.f, 0.f, 0.f};
  for (int k0 = 0; k0 < nk; k0 += 32) {
    const int kbA = k0 + (half << 4);
    v16h a, bf;
#pragma unroll
    for (int i = 0; i < 8; ++i) {
      const int k1 = kbA + i, k2 = kbA + 8 + i;
      const float v1 = Prow[(k1 < nk) ? k1 : (nk - 1)];
      const float v2 = Prow[(k2 < nk) ? k2 : (nk - 1)];
      a[i]     = (_Float16)((k1 < nk) ? v1 : 0.f);  // zero prob kills OOB terms
      a[8 + i] = (_Float16)((k2 < nk) ? v2 : 0.f);
    }
    const int kbB = k0 + (half << 4);
#pragma unroll
    for (int i = 0; i < 16; ++i) {
      const int kk = kbB + i;
      bf[i] = (_Float16)Vcol[(size_t)((kk < nk) ? kk : (nk - 1)) * D];
    }
    acc = wmma_f16(a, bf, acc);
  }
  const float* Qb = Q + (size_t)b * nq * D + h * DH + col0 + (lane & 15);
  float* Ob = O + (size_t)b * nq * D + h * DH + col0 + (lane & 15);
#pragma unroll
  for (int v = 0; v < 8; ++v) {
    const int rr = row0 + (half << 3) + v;
    if (rr < nq) {
      const size_t r = (size_t)rr * D;
      Ob[r] = Qb[r] + acc[v];
    }
  }
}

// ---------------- rowwise softmax (one wave per row) ----------------
__global__ void __launch_bounds__(32)
k_softmax(float* __restrict__ S, int nk) {
  float* p = S + (size_t)blockIdx.x * nk;
  const int lane = threadIdx.x;
  float m = -3.4e38f;
  for (int j = lane; j < nk; j += 32) m = fmaxf(m, p[j]);
#pragma unroll
  for (int off = 16; off > 0; off >>= 1) m = fmaxf(m, __shfl_xor(m, off, 32));
  float s = 0.f;
  for (int j = lane; j < nk; j += 32) { const float e = __expf(p[j] - m); p[j] = e; s += e; }
#pragma unroll
  for (int off = 16; off > 0; off >>= 1) s += __shfl_xor(s, off, 32);
  const float inv = 1.f / s;
  for (int j = lane; j < nk; j += 32) p[j] *= inv;
}

// ---------------- misc small kernels ----------------
__global__ void k_bcast(const float* __restrict__ src, float* __restrict__ dst,
                        int period, int total) {
  const int i = blockIdx.x * blockDim.x + threadIdx.x;
  if (i < total) dst[i] = src[i % period];
}

__global__ void k_take(const float* __restrict__ src, float* __restrict__ dst,
                       int k, int total) {
  const int i = blockIdx.x * blockDim.x + threadIdx.x;
  if (i >= total) return;
  const int c = i % D;
  const int r = (i / D) % k;
  const int b = i / (D * k);
  dst[i] = src[((size_t)b * KMAX + r) * D + c];
}

__global__ void k_seeds(const float* __restrict__ seed, const float* __restrict__ W,
                        const float* __restrict__ bias, float* __restrict__ seeds,
                        int kmax) {
  __shared__ float s[D];
  const int t = threadIdx.x;
  s[t] = seed[t];
  __syncthreads();
  for (int i = 0; i < kmax; ++i) {
    seeds[i * D + t] = s[t];
    float acc = bias[t];
    for (int j = 0; j < D; ++j) acc += s[j] * W[j * D + t];
    __syncthreads();
    s[t] = tanhf(acc);
    __syncthreads();
  }
}

__global__ void k_collect(const float* __restrict__ outs, const int* __restrict__ Kt,
                          float* __restrict__ vlog, float* __restrict__ sel, int k) {
  const int b = threadIdx.x;
  if (b >= BATCH) return;
  float s = 0.f;
  for (int r = 0; r < k; ++r) s += outs[((size_t)b * k + r) * OUTC];
  vlog[b * KMAX + (k - 1)] = s / (float)k;
  if (Kt[b] == k) {
    for (int r = 0; r < KMAX; ++r)
      for (int c = 0; c < OUTC; ++c)
        sel[((size_t)b * KMAX + r) * OUTC + c] =
            (r < k) ? outs[((size_t)b * k + r) * OUTC + c] : 0.f;
  }
}

__global__ void k_prep(const float* __restrict__ sel, const int* __restrict__ Kt,
                       float* __restrict__ out, float* __restrict__ pi,
                       float* __restrict__ acc) {
  const int b = threadIdx.x;
  if (b == 0) acc[0] = 0.f;
  if (b >= BATCH) return;
  const int kt = Kt[b];
  float lg[KMAX];
  for (int c = 0; c < KMAX; ++c) {
    const float* s = sel + ((size_t)b * KMAX + c) * OUTC;
    const bool valid = c < kt;
    lg[c] = valid ? s[1] : -1e10f;
    float* o = out + (size_t)b * KMAX * 4 + c * 4;
    o[0] = valid ? s[2] : 1.f;
    o[1] = valid ? s[3] : 1.f;
    o[2] = valid ? softplusf(s[4]) : 1.f;
    o[3] = valid ? softplusf(s[5]) : 1.f;
  }
  float m = lg[0];
  for (int c = 1; c < KMAX; ++c) m = fmaxf(m, lg[c]);
  float e[KMAX];
  float ssum = 0.f;
  for (int c = 0; c < KMAX; ++c) { e[c] = __expf(lg[c] - m); ssum += e[c]; }
  for (int c = 0; c < KMAX; ++c) pi[b * KMAX + c] = e[c] / ssum + 1e-10f;
}

__global__ void __launch_bounds__(256)
k_ll(const float* __restrict__ X, const float* __restrict__ params,
     const float* __restrict__ pi, float* __restrict__ acc) {
  const int idx = blockIdx.x * blockDim.x + threadIdx.x;
  float val = 0.f;
  if (idx < BATCH * NTOK) {
    const int b = idx / NTOK;
    const float x0 = X[(size_t)idx * DX + 0];
    const float x1 = X[(size_t)idx * DX + 1];
    const float* pm = params + (size_t)b * KMAX * 4;
    float l[KMAX];
    float m = -3.4e38f;
    for (int j = 0; j < KMAX; ++j) {
      const float mu0 = pm[j * 4 + 0], mu1 = pm[j * 4 + 1];
      const float sg0 = pm[j * 4 + 2], sg1 = pm[j * 4 + 3];
      const float z0 = (x0 - mu0) / sg0, z1 = (x1 - mu1) / sg1;
      const float lp = -0.5f * (z0 * z0 + z1 * z1) - __logf(sg0) - __logf(sg1)
                       - 1.8378770664093453f;
      l[j] = lp + __logf(pi[b * KMAX + j]);
      m = fmaxf(m, l[j]);
    }
    float ssum = 0.f;
    for (int j = 0; j < KMAX; ++j) ssum += __expf(l[j] - m);
    val = m + __logf(ssum);
  }
  __shared__ float red[256];
  red[threadIdx.x] = val;
  __syncthreads();
  for (int s = 128; s > 0; s >>= 1) {
    if (threadIdx.x < s) red[threadIdx.x] += red[threadIdx.x + s];
    __syncthreads();
  }
  if (threadIdx.x == 0) atomicAdd(acc, red[0]);
}

__global__ void k_bc(const float* __restrict__ vlog, const int* __restrict__ Kt,
                     const float* __restrict__ acc, float* __restrict__ out) {
  __shared__ float red[64];
  const int b = threadIdx.x;  // blockDim == 64 == BATCH
  float s = 0.f;
  const int kt = Kt[b];
  float log_c = 0.f;
  for (int c = 0; c < KMAX; ++c) {
    const float v = vlog[b * KMAX + c];
    const float sig = 1.f / (1.f + __expf(-v));
    log_c += __logf(sig + 1e-10f);
    const float cl = log_c - __logf(1.f - __expf(log_c) + 1e-10f);
    const float t = (c < kt - 1) ? 1.f : 0.f;
    s += fmaxf(cl, 0.f) - cl * t + log1pf(__expf(-fabsf(cl)));
  }
  red[b] = s;
  __syncthreads();
  for (int st = 32; st > 0; st >>= 1) {
    if (b < st) red[b] += red[b + st];
    __syncthreads();
  }
  if (b == 0) {
    out[BATCH * KMAX * 4 + 0] = acc[0] / (float)(BATCH * NTOK);
    out[BATCH * KMAX * 4 + 1] = red[0] / (float)(BATCH * KMAX);
  }
}

// ---------------- host orchestration ----------------
struct LinP { const float* w; const float* b; _Float16* wf; int K, N; };
struct MabP { LinP q, k, v, o; };

static void gemm(hipStream_t s, const float* X, const LinP& L, const float* resid,
                 float* Y, int M, int op) {
  const int N = L.N, K = L.K;
  if ((M % 16 == 0) && (K == D) && (N % 64 == 0))
    k_gemm_f128<<<dim3(M / 16, N / 64), 32, 0, s>>>(X, L.wf, L.b, resid, Y, M, N, op);
  else if ((M % 16 == 0) && (K % 32 == 0) && (N % 64 == 0))
    k_gemm_f<<<dim3(M / 16, N / 64), 32, 0, s>>>(X, L.wf, L.b, resid, Y, M, N, K, op);
  else
    k_gemm_e<<<dim3((M + 15) / 16, (N + 15) / 16), 32, 0, s>>>(X, L.wf, L.b, resid, Y, M, N, K, op);
}

static void run_mab(hipStream_t s, const MabP& m,
                    const float* Xq, int nq,
                    const float* Y, int nk,
                    float* Qb, float* Kb, float* Vb, float* Sb, float* Out) {
  gemm(s, Xq, m.q, nullptr, Qb, BATCH * nq, 0);
  gemm(s, Y,  m.k, nullptr, Kb, BATCH * nk, 0);
  gemm(s, Y,  m.v, nullptr, Vb, BATCH * nk, 0);
  const float scale = 0.08838834764831845f;  // 1/sqrt(D)
  if ((nq % 16 == 0) && (nk % 16 == 0))
    k_scores_f<<<dim3(nq / 16, nk / 16, BATCH * HEADS), 32, 0, s>>>(Qb, Kb, Sb, nq, nk, scale);
  else
    k_scores_c<<<dim3((nq + 15) / 16, (nk + 15) / 16, BATCH * HEADS), 32, 0, s>>>(Qb, Kb, Sb, nq, nk, scale);
  k_softmax<<<dim3(BATCH * HEADS * nq), 32, 0, s>>>(Sb, nk);
  if ((nq % 16 == 0) && (nk % 32 == 0))
    k_av_f<<<dim3(nq / 16, DH / 16, BATCH * HEADS), 32, 0, s>>>(Sb, Vb, Qb, Qb, nq, nk);
  else
    k_av_c<<<dim3((nq + 15) / 16, DH / 16, BATCH * HEADS), 32, 0, s>>>(Sb, Vb, Qb, Qb, nq, nk);
  gemm(s, Qb, m.o, Qb, Out, BATCH * nq, 1);  // Out = O + relu(O@Wo+bo)
}

extern "C" void kernel_launch(void* const* d_in, const int* in_sizes, int n_in,
                              void* d_out, int out_size, void* d_ws, size_t ws_size,
                              hipStream_t stream) {
  (void)in_sizes; (void)n_in; (void)out_size; (void)ws_size;
  // --- flatten params in setup_inputs() insertion order ---
  int idx = 0;
  auto nf   = [&]() -> const float* { return (const float*)d_in[idx++]; };
  auto nlin = [&](int K, int N) { LinP l; l.w = nf(); l.b = nf(); l.wf = nullptr; l.K = K; l.N = N; return l; };
  auto nmab = [&](int dq, int dk) {
    MabP m; m.q = nlin(dq, D); m.k = nlin(dk, D); m.v = nlin(dk, D); m.o = nlin(D, D); return m;
  };

  const float* Xp = nf();                      // X (B,N,DX)
  const float* Ivec[4]; MabP mab0[4], mab1[4];
  for (int L = 0; L < 4; ++L) {
    const int din = (L == 0) ? DX : D;
    Ivec[L] = nf();
    mab0[L] = nmab(D, din);    // mab0: dq=D, dk=din
    mab1[L] = nmab(din, D);    // mab1: dq=din, dk=D
  }
  const float* seed = nf();
  const float* seed_fc_w = nf();
  const float* seed_fc_b = nf();
  MabP amab = nmab(D, D);
  MabP sab0 = nmab(D, D), sab1 = nmab(D, D);
  LinP fc = nlin(D, OUTC);
  const int* Ktrue = (const int*)d_in[idx++];  // K_true (K_max ignored)

  // --- workspace bump allocator (~406 MB) ---
  char* base = (char*)d_ws;
  size_t off = 0;
  auto allocB = [&](size_t bytes) { char* p = base + off; off += (bytes + 255) & ~(size_t)255; return p; };
  auto allocF = [&](size_t n) { return (float*)allocB(n * sizeof(float)); };
  auto allocH = [&](size_t n) { return (_Float16*)allocB(n * sizeof(_Float16)); };

  const size_t TOK = (size_t)BATCH * NTOK * D;
  float* HA = allocF(TOK);
  float* HB = allocF(TOK);
  float* Qb = allocF(TOK);
  float* Kb = allocF(TOK);
  float* Vb = allocF(TOK);
  float* Sb = allocF((size_t)BATCH * HEADS * NTOK * NI);
  float* Hind   = allocF((size_t)BATCH * NI * D);
  float* Ib     = allocF((size_t)BATCH * NI * D);
  float* seeds  = allocF((size_t)KMAX * D);
  float* seedsB = allocF((size_t)BATCH * KMAX * D);
  float* Zfull  = allocF((size_t)BATCH * KMAX * D);
  float* Z1     = allocF((size_t)BATCH * KMAX * D);
  float* Z2     = allocF((size_t)BATCH * KMAX * D);
  float* outs   = allocF((size_t)BATCH * KMAX * OUTC);
  float* sel    = allocF((size_t)BATCH * KMAX * OUTC);
  float* vlog   = allocF((size_t)BATCH * KMAX);
  float* pib    = allocF((size_t)BATCH * KMAX);
  float* accb   = allocF(16);

  // --- one-time weight prep into fragment-native f16 ---
  auto prep = [&](LinP& l) {
    const int Kpad = (l.K + 31) & ~31;
    const int total = Kpad * l.N;
    l.wf = allocH((size_t)total);
    k_wprep<<<(total + 255) / 256, 256, 0, stream>>>(l.w, l.wf, l.K, l.N, total);
  };
  auto prepMab = [&](MabP& m) { prep(m.q); prep(m.k); prep(m.v); prep(m.o); };
  for (int L = 0; L < 4; ++L) { prepMab(mab0[L]); prepMab(mab1[L]); }
  prepMab(amab); prepMab(sab0); prepMab(sab1); prep(fc);

  // --- encoder: 4 ISAB blocks ---
  const float* Hin = Xp;
  int pb = 0;
  float* bufs[2] = {HA, HB};
  for (int L = 0; L < 4; ++L) {
    const int tot = BATCH * NI * D;
    k_bcast<<<(tot + 255) / 256, 256, 0, stream>>>(Ivec[L], Ib, NI * D, tot);
    run_mab(stream, mab0[L], Ib, NI, Hin, NTOK, Qb, Kb, Vb, Sb, Hind);
    run_mab(stream, mab1[L], Hin, NTOK, Hind, NI, Qb, Kb, Vb, Sb, bufs[pb]);
    Hin = bufs[pb]; pb ^= 1;
  }
  const float* H4 = Hin;

  // --- APMA: seed scan once; pooling MAB once with KMAX rows
  //     (rows pass through a MAB independently, so Z0(k) is a row prefix) ---
  k_seeds<<<1, D, 0, stream>>>(seed, seed_fc_w, seed_fc_b, seeds, KMAX);
  {
    const int tot = BATCH * KMAX * D;
    k_bcast<<<(tot + 255) / 256, 256, 0, stream>>>(seeds, seedsB, KMAX * D, tot);
  }
  run_mab(stream, amab, seedsB, KMAX, H4, NTOK, Qb, Kb, Vb, Sb, Zfull);

  // --- decoder loop: SABs + fc on k-row prefixes ---
  for (int k = 1; k <= KMAX; ++k) {
    const int tot = BATCH * k * D;
    k_take<<<(tot + 255) / 256, 256, 0, stream>>>(Zfull, Z1, k, tot);
    run_mab(stream, sab0, Z1, k, Z1, k, Qb, Kb, Vb, Sb, Z2);
    run_mab(stream, sab1, Z2, k, Z2, k, Qb, Kb, Vb, Sb, Z1);
    gemm(stream, Z1, fc, nullptr, outs, BATCH * k, 0);
    k_collect<<<1, 64, 0, stream>>>(outs, Ktrue, vlog, sel, k);
  }

  // --- outputs and losses ---
  float* outp = (float*)d_out;
  k_prep<<<1, 64, 0, stream>>>(sel, Ktrue, outp, pib, accb);
  k_ll<<<(BATCH * NTOK + 255) / 256, 256, 0, stream>>>(Xp, outp, pib, accb);
  k_bc<<<1, 64, 0, stream>>>(vlog, Ktrue, accb, outp);
}
